// GraphModel_57578331570178
// MI455X (gfx1250) — matrix operand order repaired and checked by reading, not compiled, not run on hardware
//
#include <hip/hip_runtime.h>
#include <hip/hip_bf16.h>

// ---------------------------------------------------------------------------
// Problem constants (match reference)
// ---------------------------------------------------------------------------
constexpr int NN   = 65536;   // nodes
constexpr int EE   = 262144;  // edges
constexpr int BGn  = 256;     // graphs
constexpr int SS   = 256;     // nodes per graph
constexpr int CC   = 128;     // channels
constexpr int LL   = 4;       // layers
constexpr int DST  = 16;      // d_state
constexpr int DTR  = 8;       // dt rank
constexpr int XPD  = DTR + 2 * DST;  // 40

typedef __attribute__((ext_vector_type(16))) _Float16 v16h;
typedef __attribute__((ext_vector_type(8)))  _Float16 v8h;
typedef __attribute__((ext_vector_type(8)))  float    v8f;

// ---------------------------------------------------------------------------
// WMMA GEMM:  out[M,Nc] = op(A[M,K] @ B[K,Nc] [+ bias] [+ res1 [+ res2]])
// f32 in/out, f16 MAC via v_wmma_f32_16x16x32_f16 (f32 accumulate).
//
// Block = 256 threads = 8 waves; block tile 128x128 (BK=32).
// Wave grid 4(M) x 2(N); each wave owns 32x64 = 8 WMMA tiles per k-step
// (2 A-fragments reused across 4 B-fragments).
// M multiple of 128, K multiple of 32.  GUARD=true handles Nc not a
// multiple of 128 (single per-thread column predicate; used only for Nc=40).
// BIAS / NRES / RELU are compile-time so the epilogue is straight-line code.
// ---------------------------------------------------------------------------
template <bool GUARD, bool BIAS, int NRES, bool RELU>
__global__ __launch_bounds__(256) void gemm_wmma(
    const float* __restrict__ A, const float* __restrict__ B,
    const float* __restrict__ bias, const float* __restrict__ res1,
    const float* __restrict__ res2, float* __restrict__ out,
    int K, int Nc)
{
  const int tid  = threadIdx.x;
  const int lane = tid & 31;
  const int w    = tid >> 5;
  const int wm   = w >> 1;          // 0..3 : M offset wm*32
  const int wn   = w & 1;           // 0..1 : N offset wn*64
  const int m0   = blockIdx.x * 128;
  const int n0   = blockIdx.y * 128;

  __shared__ __align__(16) _Float16 As[128][40];   // [m][k], padded stride
  __shared__ __align__(16) _Float16 Bs[128][40];   // [n][k], B transposed

  v8f acc[2][4];
#pragma unroll
  for (int i = 0; i < 2; ++i)
#pragma unroll
    for (int j = 0; j < 4; ++j) acc[i][j] = {};

  // ---- cooperative staging indices
  // A: thread owns 16 consecutive k of one row (64B contiguous global read).
  const int ar  = tid >> 1;            // 0..127
  const int ak  = (tid & 1) * 16;      // 0 or 16
  // B: thread owns one column n and 16 consecutive k (lane-coalesced b32
  //    global reads; contiguous 16B LDS stores into transposed tile).
  const int bn  = tid & 127;           // 0..127
  const int bko = (tid >> 7) * 16;     // 0 or 16
  const bool bvalid = !GUARD || (n0 + bn < Nc);

  const int r    = lane & 15;
  const int half = lane >> 4;

  for (int k0 = 0; k0 < K; k0 += 32) {
    __syncthreads();
    // ---- stage A tile (f32 -> f16)
    {
      const float* ap = A + (size_t)(m0 + ar) * K + (k0 + ak);
      float4 f0 = *(const float4*)ap;
      float4 f1 = *(const float4*)(ap + 4);
      float4 f2 = *(const float4*)(ap + 8);
      float4 f3 = *(const float4*)(ap + 12);
      v8h h0, h1;
      h0[0] = (_Float16)f0.x; h0[1] = (_Float16)f0.y;
      h0[2] = (_Float16)f0.z; h0[3] = (_Float16)f0.w;
      h0[4] = (_Float16)f1.x; h0[5] = (_Float16)f1.y;
      h0[6] = (_Float16)f1.z; h0[7] = (_Float16)f1.w;
      h1[0] = (_Float16)f2.x; h1[1] = (_Float16)f2.y;
      h1[2] = (_Float16)f2.z; h1[3] = (_Float16)f2.w;
      h1[4] = (_Float16)f3.x; h1[5] = (_Float16)f3.y;
      h1[6] = (_Float16)f3.z; h1[7] = (_Float16)f3.w;
      *(v8h*)&As[ar][ak]     = h0;
      *(v8h*)&As[ar][ak + 8] = h1;
    }
    // ---- stage B tile transposed
    {
      const float* bp = B + (size_t)(k0 + bko) * Nc + (n0 + bn);
      v8h h0, h1;
#pragma unroll
      for (int j = 0; j < 8; ++j) {
        float v0 = bvalid ? bp[(size_t)j * Nc]       : 0.0f;
        float v1 = bvalid ? bp[(size_t)(j + 8) * Nc] : 0.0f;
        h0[j] = (_Float16)v0;
        h1[j] = (_Float16)v1;
      }
      *(v8h*)&Bs[bn][bko]     = h0;
      *(v8h*)&Bs[bn][bko + 8] = h1;
    }
    __syncthreads();

    // ---- A fragments (rows wm*32 + i*16 + r)
    v16h a[2];
#pragma unroll
    for (int i = 0; i < 2; ++i) {
      v8h alo = *(const v8h*)&As[wm * 32 + i * 16 + r][8 * half];
      v8h ahi = *(const v8h*)&As[wm * 32 + i * 16 + r][16 + 8 * half];
#pragma unroll
      for (int q = 0; q < 8; ++q) { a[i][q] = alo[q]; a[i][8 + q] = ahi[q]; }
    }
    // ---- B fragments (cols wn*64 + j*16 + r); live range limited to one j
#pragma unroll
    for (int j = 0; j < 4; ++j) {
      const int nb = wn * 64 + j * 16 + r;
      v8h blo = *(const v8h*)&Bs[nb][8 * half];
      v8h bhi = *(const v8h*)&Bs[nb][16 + 8 * half];
      v16h b;
#pragma unroll
      for (int q = 0; q < 8; ++q) { b[q] = blo[q]; b[8 + q] = bhi[q]; }
      acc[0][j] = __builtin_amdgcn_wmma_f32_16x16x32_f16(false, a[0], false, b,
                                                         (short)0, acc[0][j],
                                                         false, false);
      acc[1][j] = __builtin_amdgcn_wmma_f32_16x16x32_f16(false, a[1], false, b,
                                                         (short)0, acc[1][j],
                                                         false, false);
    }
  }

  // ---- epilogue (straight-line; D layout: lane n=r, VGPR rr -> m=rr+8*half)
#pragma unroll
  for (int i = 0; i < 2; ++i) {
#pragma unroll
    for (int j = 0; j < 4; ++j) {
      const int ng = n0 + wn * 64 + j * 16 + r;
      if (!GUARD || ng < Nc) {
        const float bv = BIAS ? bias[ng] : 0.0f;
        const int mg0 = m0 + wm * 32 + i * 16 + 8 * half;
        const size_t base = (size_t)mg0 * Nc + ng;
#pragma unroll
        for (int rr = 0; rr < 8; ++rr) {
          const size_t off = base + (size_t)rr * Nc;
          float v = acc[i][j][rr];
          if (BIAS) v += bv;
          if (NRES >= 1) v += res1[off];
          if (NRES >= 2) v += res2[off];
          if (RELU) v = fmaxf(v, 0.0f);
          out[off] = v;
        }
      }
    }
  }
}

// ---------------------------------------------------------------------------
// BatchNorm stats: per-dim sum / sumsq over N rows. grid = 20 blocks.
// ---------------------------------------------------------------------------
__global__ __launch_bounds__(256) void bn_stats(const float* __restrict__ pe,
                                                float* __restrict__ stats)
{
  const int d = blockIdx.x, tid = threadIdx.x;
  __shared__ float s1[256], s2[256];
  float a = 0.0f, b = 0.0f;
  for (int rIdx = tid; rIdx < NN; rIdx += 256) {
    float v = pe[(size_t)rIdx * 20 + d];
    a += v; b += v * v;
  }
  s1[tid] = a; s2[tid] = b;
  __syncthreads();
  for (int off = 128; off; off >>= 1) {
    if (tid < off) { s1[tid] += s1[tid + off]; s2[tid] += s2[tid + off]; }
    __syncthreads();
  }
  if (tid == 0) { stats[d] = s1[0]; stats[20 + d] = s2[0]; }
}

// ---------------------------------------------------------------------------
// Embedding: x[i] = concat(node_emb[tok[i]], BN(pe[i]) @ pe_lin + b)
// ---------------------------------------------------------------------------
__global__ __launch_bounds__(128) void embed_kernel(
    const float* __restrict__ pe, const int* __restrict__ tok,
    const float* __restrict__ nemb, const float* __restrict__ pw,
    const float* __restrict__ pb, const float* __restrict__ gamma,
    const float* __restrict__ beta, const float* __restrict__ stats,
    float* __restrict__ x)
{
  const int i = blockIdx.x, c = threadIdx.x;
  __shared__ float pn[20];
  if (c < 20) {
    const float inv_n = 1.0f / (float)NN;
    float mu  = stats[c] * inv_n;
    float var = stats[20 + c] * inv_n - mu * mu;
    float inv = rsqrtf(var + 1e-5f);
    pn[c] = (pe[(size_t)i * 20 + c] - mu) * inv * gamma[c] + beta[c];
  }
  __syncthreads();
  if (c < 120) {
    x[(size_t)i * CC + c] = nemb[tok[i] * 120 + c];
  } else {
    const int o = c - 120;
    float a = pb[o];
#pragma unroll
    for (int k = 0; k < 20; ++k) a += pn[k] * pw[k * 8 + o];
    x[(size_t)i * CC + c] = a;
  }
}

// ---------------------------------------------------------------------------
// GINE edge scatter: h[dst] += relu(x[src] + edge_emb[attr]) (h pre-init to x)
// ---------------------------------------------------------------------------
__global__ __launch_bounds__(128) void gine_edge(
    const float* __restrict__ x, const float* __restrict__ ee,
    const int* __restrict__ ei, const int* __restrict__ ea,
    float* __restrict__ h)
{
  const int e = blockIdx.x, c = threadIdx.x;
  const int sn = ei[e];
  const int dn = ei[EE + e];
  const int at = ea[e];
  float v = x[(size_t)sn * CC + c] + ee[at * CC + c];
  v = fmaxf(v, 0.0f);
  __hip_atomic_fetch_add(&h[(size_t)dn * CC + c], v,
                         __ATOMIC_RELAXED, __HIP_MEMORY_SCOPE_AGENT);
}

// ---------------------------------------------------------------------------
// Causal depthwise conv (width 4, per graph) + SiLU.  xz: [N, 2C], xi = cols 0..C
// ---------------------------------------------------------------------------
__global__ __launch_bounds__(128) void dwconv_silu(
    const float* __restrict__ xz, const float* __restrict__ cw,
    const float* __restrict__ cb, float* __restrict__ xc)
{
  const int i = blockIdx.x, c = threadIdx.x;
  const int s = i & (SS - 1);
  float acc = cb[c];
#pragma unroll
  for (int j = 0; j < 4; ++j) {
    const int d = s - (3 - j);
    if (d >= 0) acc += cw[c * 4 + j] * xz[(size_t)(i - (3 - j)) * (2 * CC) + c];
  }
  xc[(size_t)i * CC + c] = acc / (1.0f + __expf(-acc));  // silu
}

// ---------------------------------------------------------------------------
// Fused selective scan: dt-softplus + recurrence + y = ys + Dp*xc, g = y*silu(z)
// grid = BG graphs, 128 threads = channels; dtr/B/C staged per step in LDS.
// ---------------------------------------------------------------------------
__global__ __launch_bounds__(128) void mamba_scan(
    const float* __restrict__ xp,  const float* __restrict__ xc,
    const float* __restrict__ xz,  const float* __restrict__ dtw,
    const float* __restrict__ dtb, const float* __restrict__ Alog,
    const float* __restrict__ Dpv, float* __restrict__ g)
{
  const int b = blockIdx.x, c = threadIdx.x;
  __shared__ float sXP[XPD];   // [0,8)=dtr  [8,24)=B  [24,40)=C
  float Aneg[DST];
#pragma unroll
  for (int n = 0; n < DST; ++n) Aneg[n] = -__expf(Alog[c * DST + n]);
  float dw[DTR];
#pragma unroll
  for (int rIdx = 0; rIdx < DTR; ++rIdx) dw[rIdx] = dtw[rIdx * CC + c];
  const float db  = dtb[c];
  const float dpc = Dpv[c];
  float h[DST];
#pragma unroll
  for (int n = 0; n < DST; ++n) h[n] = 0.0f;

  for (int s = 0; s < SS; ++s) {
    const int row = b * SS + s;
    __syncthreads();
    if (c < XPD) sXP[c] = xp[(size_t)row * XPD + c];
    __syncthreads();
    float da = db;
#pragma unroll
    for (int rIdx = 0; rIdx < DTR; ++rIdx) da += sXP[rIdx] * dw[rIdx];
    const float dt  = (da > 20.0f) ? da : log1pf(__expf(da));
    const float xv  = xc[(size_t)row * CC + c];
    const float dtx = dt * xv;
    float y = 0.0f;
#pragma unroll
    for (int n = 0; n < DST; ++n) {
      const float dA = __expf(dt * Aneg[n]);
      h[n] = dA * h[n] + dtx * sXP[DTR + n];
      y += h[n] * sXP[DTR + DST + n];
    }
    const float zv = xz[(size_t)row * (2 * CC) + CC + c];
    const float sz = zv / (1.0f + __expf(-zv));
    g[(size_t)row * CC + c] = (y + dpc * xv) * sz;
  }
}

// ---------------------------------------------------------------------------
// Pool (global add pool over S contiguous rows) and final MLP head
// ---------------------------------------------------------------------------
__global__ __launch_bounds__(128) void pool_kernel(const float* __restrict__ x,
                                                   float* __restrict__ pooled)
{
  const int b = blockIdx.x, c = threadIdx.x;
  const float* p = x + (size_t)b * SS * CC + c;
  float a = 0.0f;
  for (int s = 0; s < SS; ++s) a += p[(size_t)s * CC];
  pooled[b * CC + c] = a;
}

__global__ __launch_bounds__(64) void head_kernel(
    const float* __restrict__ pooled,
    const float* __restrict__ w1, const float* __restrict__ b1,
    const float* __restrict__ w2, const float* __restrict__ b2,
    const float* __restrict__ w3, const float* __restrict__ b3,
    float* __restrict__ out)
{
  const int b = blockIdx.x, t = threadIdx.x;
  __shared__ float pr[128], h1[64], h2[32];
  pr[t]      = pooled[b * CC + t];
  pr[t + 64] = pooled[b * CC + t + 64];
  __syncthreads();
  float a = b1[t];
  for (int k = 0; k < 128; ++k) a += pr[k] * w1[k * 64 + t];
  h1[t] = fmaxf(a, 0.0f);
  __syncthreads();
  if (t < 32) {
    float a2 = b2[t];
    for (int k = 0; k < 64; ++k) a2 += h1[k] * w2[k * 32 + t];
    h2[t] = fmaxf(a2, 0.0f);
  }
  __syncthreads();
  if (t == 0) {
    float a3 = b3[0];
    for (int k = 0; k < 32; ++k) a3 += h2[k] * w3[k];
    out[b] = a3;
  }
}

__global__ __launch_bounds__(256) void copy4(const float4* __restrict__ s,
                                             float4* __restrict__ d, int n)
{
  const int i = blockIdx.x * 256 + threadIdx.x;
  if (i < n) d[i] = s[i];
}

// ---------------------------------------------------------------------------
// Host orchestration
// ---------------------------------------------------------------------------
extern "C" void kernel_launch(void* const* d_in, const int* in_sizes, int n_in,
                              void* d_out, int out_size, void* d_ws, size_t ws_size,
                              hipStream_t stream)
{
  (void)in_sizes; (void)n_in; (void)out_size; (void)ws_size;

  const float* pe    = (const float*)d_in[0];
  const int*   tok   = (const int*)d_in[1];
  const int*   eidx  = (const int*)d_in[2];
  const int*   eattr = (const int*)d_in[3];
  /* d_in[4] = batch: graphs are node-contiguous -> implicit */
  const float* nemb  = (const float*)d_in[5];
  const float* pew   = (const float*)d_in[6];
  const float* peb   = (const float*)d_in[7];
  const float* gam   = (const float*)d_in[8];
  const float* bet   = (const float*)d_in[9];
  const float* ee    = (const float*)d_in[10];
  const float* gw1   = (const float*)d_in[11];
  const float* gb1   = (const float*)d_in[12];
  const float* gw2   = (const float*)d_in[13];
  const float* gb2   = (const float*)d_in[14];
  const float* inw   = (const float*)d_in[15];
  const float* cw    = (const float*)d_in[16];
  const float* cb    = (const float*)d_in[17];
  const float* xpw   = (const float*)d_in[18];
  const float* dtw   = (const float*)d_in[19];
  const float* dtb   = (const float*)d_in[20];
  const float* Alog  = (const float*)d_in[21];
  const float* Dpv   = (const float*)d_in[22];
  const float* ow    = (const float*)d_in[23];
  const float* mw1   = (const float*)d_in[24];
  const float* mb1   = (const float*)d_in[25];
  const float* mw2   = (const float*)d_in[26];
  const float* mb2   = (const float*)d_in[27];
  const float* fw1   = (const float*)d_in[28];
  const float* fb1   = (const float*)d_in[29];
  const float* fw2   = (const float*)d_in[30];
  const float* fb2   = (const float*)d_in[31];
  const float* fw3   = (const float*)d_in[32];
  const float* fb3   = (const float*)d_in[33];

  // workspace layout (floats)
  float* W = (float*)d_ws;
  const size_t NC = (size_t)NN * CC;          // 8,388,608
  float* xbuf   = W;                          // x                 [N,C]
  float* hbuf   = W + 1 * NC;                 // h / x2            [N,C]
  float* t1buf  = W + 2 * NC;                 // t1 / g            [N,C]
  float* hlbuf  = W + 3 * NC;                 // h_local           [N,C]
  float* xzbuf  = W + 4 * NC;                 // xz / mlp-hidden   [N,2C]
  float* xcbuf  = W + 6 * NC;                 // xc                [N,C]
  float* xpbuf  = W + 7 * NC;                 // x_proj out        [N,40]
  float* stats  = W + 7 * NC + (size_t)NN * XPD;  // 40
  float* pooled = stats + 64;                 // [BG,C]

  const dim3 blk(256);
  const dim3 g1(NN / 128, 1);   // Nc <= 128
  const dim3 g2(NN / 128, 2);   // Nc == 256

  // ---- input embedding
  bn_stats<<<20, 256, 0, stream>>>(pe, stats);
  embed_kernel<<<NN, 128, 0, stream>>>(pe, tok, nemb, pew, peb, gam, bet, stats, xbuf);

  for (int l = 0; l < LL; ++l) {
    const float* gw1l = gw1 + (size_t)l * CC * CC;
    const float* gb1l = gb1 + (size_t)l * CC;
    const float* gw2l = gw2 + (size_t)l * CC * CC;
    const float* gb2l = gb2 + (size_t)l * CC;
    const float* inwl = inw + (size_t)l * CC * 2 * CC;
    const float* cwl  = cw  + (size_t)l * CC * 4;
    const float* cbl  = cb  + (size_t)l * CC;
    const float* xpwl = xpw + (size_t)l * CC * XPD;
    const float* dtwl = dtw + (size_t)l * DTR * CC;
    const float* dtbl = dtb + (size_t)l * CC;
    const float* All  = Alog + (size_t)l * CC * DST;
    const float* Dpl  = Dpv + (size_t)l * CC;
    const float* owl  = ow  + (size_t)l * CC * CC;
    const float* mw1l = mw1 + (size_t)l * CC * 2 * CC;
    const float* mb1l = mb1 + (size_t)l * 2 * CC;
    const float* mw2l = mw2 + (size_t)l * 2 * CC * CC;
    const float* mb2l = mb2 + (size_t)l * CC;

    // GINE: h = x + sum_{edges} relu(x[src] + e)
    copy4<<<(int)(NC / 4 / 256), 256, 0, stream>>>((const float4*)xbuf,
                                                   (float4*)hbuf, (int)(NC / 4));
    gine_edge<<<EE, 128, 0, stream>>>(xbuf, ee, eidx, eattr, hbuf);

    // t1 = relu(h @ W1 + b1)
    gemm_wmma<false, true, 0, true><<<g1, blk, 0, stream>>>(
        hbuf, gw1l, gb1l, nullptr, nullptr, t1buf, CC, CC);
    // h_local = t1 @ W2 + b2 + x
    gemm_wmma<false, true, 1, false><<<g1, blk, 0, stream>>>(
        t1buf, gw2l, gb2l, xbuf, nullptr, hlbuf, CC, CC);

    // Mamba global branch
    // xz = x @ in_proj
    gemm_wmma<false, false, 0, false><<<g2, blk, 0, stream>>>(
        xbuf, inwl, nullptr, nullptr, nullptr, xzbuf, CC, 2 * CC);
    dwconv_silu<<<NN, 128, 0, stream>>>(xzbuf, cwl, cbl, xcbuf);
    // xp = xc @ x_proj  (Nc = 40, guarded)
    gemm_wmma<true, false, 0, false><<<g1, blk, 0, stream>>>(
        xcbuf, xpwl, nullptr, nullptr, nullptr, xpbuf, CC, XPD);
    mamba_scan<<<BGn, 128, 0, stream>>>(xpbuf, xcbuf, xzbuf, dtwl, dtbl,
                                        All, Dpl, t1buf);                 // g
    // x2 = g @ out_w + h_local + x
    gemm_wmma<false, false, 2, false><<<g1, blk, 0, stream>>>(
        t1buf, owl, nullptr, hlbuf, xbuf, hbuf, CC, CC);

    // MLP
    gemm_wmma<false, true, 0, true><<<g2, blk, 0, stream>>>(
        hbuf, mw1l, mb1l, nullptr, nullptr, xzbuf, CC, 2 * CC);
    gemm_wmma<false, true, 0, false><<<g1, blk, 0, stream>>>(
        xzbuf, mw2l, mb2l, nullptr, nullptr, xbuf, 2 * CC, CC);
  }

  // pooling + head
  pool_kernel<<<BGn, 128, 0, stream>>>(xbuf, pooled);
  head_kernel<<<BGn, 64, 0, stream>>>(pooled, fw1, fb1, fw2, fb2, fw3, fb3,
                                      (float*)d_out);
}
